// GQA_Attention_17282948399107
// MI455X (gfx1250) — matrix operand order repaired and checked by reading, not compiled, not run on hardware
//
#include <hip/hip_runtime.h>

// ---------------- problem constants ----------------
#define S_     2048
#define E_     2048
#define HQ_    32
#define HKV_   8
#define DH_    64
#define CACHE_ 2048
#define T_     4096          // CACHE + S
#define SCALE_ 0.125f        // D^-0.5
#define K2_    0.18033688f   // SCALE_ * log2(e)

typedef _Float16 h16;
typedef __attribute__((ext_vector_type(16))) _Float16     v16h;
typedef __attribute__((ext_vector_type(8)))  float        v8f;
typedef __attribute__((ext_vector_type(4)))  unsigned int v4u;

union FragU { v16h v; v4u q[2]; };

#define WMMA_F16(a, b, c) \
  __builtin_amdgcn_wmma_f32_16x16x32_f16(false, (a), false, (b), (short)0, (c), false, false)

// Build a 16-element f16 fragment from two 16B-aligned 8-element chunks.
__device__ __forceinline__ v16h frag2(const h16* p0, const h16* p1) {
  FragU r;
  r.q[0] = *(const v4u*)p0;
  r.q[1] = *(const v4u*)p1;
  return r.v;
}

// A operand (16x32, row-major [M][K], stride lda): lane holds row (lane&15),
// K chunks [kb, kb+8) and [kb+16, kb+24), kb = (lane&16) ? 8 : 0  (ISA 7.12.2)
__device__ __forceinline__ v16h load_a_frag(const h16* A, int lda, int m0, int k0) {
  int lane = threadIdx.x & 31;
  const h16* p = A + (size_t)(m0 + (lane & 15)) * lda + k0 + ((lane & 16) ? 8 : 0);
  return frag2(p, p + 16);
}

// B operand from Bt stored [N][K] row-major (stride ldb): lane holds col (lane&15),
// K chunks [kb, kb+8) and [kb+16, kb+24), kb = (lane&16) ? 8 : 0
__device__ __forceinline__ v16h load_b_frag(const h16* Bt, int ldb, int n0, int k0) {
  int lane = threadIdx.x & 31;
  const h16* p = Bt + (size_t)(n0 + (lane & 15)) * ldb + k0 + ((lane & 16) ? 8 : 0);
  return frag2(p, p + 16);
}

// ---------------- prep kernels (bandwidth-trivial) ----------------
__global__ void cast_f16_kernel(const float* __restrict__ in, h16* __restrict__ out, int n) {
  for (int i = blockIdx.x * blockDim.x + threadIdx.x; i < n; i += gridDim.x * blockDim.x)
    out[i] = (h16)in[i];
}

// in fp32 [R][C] -> out f16 [C][R]
__global__ void transpose_cast_kernel(const float* __restrict__ in, h16* __restrict__ out,
                                      int R, int C) {
  int n = R * C;
  for (int i = blockIdx.x * blockDim.x + threadIdx.x; i < n; i += gridDim.x * blockDim.x) {
    int r = i / C, c = i - r * C;
    out[(size_t)c * R + r] = (h16)in[i];
  }
}

// prev_k/prev_v [CACHE][HKV*DH] -> new_k/new_v fp32 (t<CACHE part) + f16 attn layouts:
//   Kb[h][t][d]  (contiguous d)   Vt[h][d][t] == flat [c][t] with c = h*64+d
__global__ void prep_cache_kernel(const float* __restrict__ pk, const float* __restrict__ pv,
                                  float* __restrict__ newk, float* __restrict__ newv,
                                  h16* __restrict__ Kb, h16* __restrict__ Vt) {
  int n = CACHE_ * HKV_ * DH_;
  for (int i = blockIdx.x * blockDim.x + threadIdx.x; i < n; i += gridDim.x * blockDim.x) {
    int t = i >> 9, c = i & 511;            // 512 = HKV*DH
    float kv = pk[i];
    newk[i] = kv;
    Kb[((size_t)(c >> 6) * T_ + t) * DH_ + (c & 63)] = (h16)kv;
    float vv = pv[i];
    newv[i] = vv;
    Vt[(size_t)c * T_ + t] = (h16)vv;
  }
}

// ---------------- generic wave-level 32x32 WMMA GEMM ----------------
// C[32x32] = A[M][K] (f16 row-major) x Bt[N][K]^T, lda = ldb = K
__device__ __forceinline__ void wave_gemm(const h16* A, const h16* Bt, int K,
                                          int m0, int n0, v8f c[2][2]) {
  v8f z = {};
  c[0][0] = z; c[0][1] = z; c[1][0] = z; c[1][1] = z;
  for (int k0 = 0; k0 < K; k0 += 32) {
    v16h a0 = load_a_frag(A, K, m0,      k0);
    v16h a1 = load_a_frag(A, K, m0 + 16, k0);
    v16h b0 = load_b_frag(Bt, K, n0,      k0);
    v16h b1 = load_b_frag(Bt, K, n0 + 16, k0);
    c[0][0] = WMMA_F16(a0, b0, c[0][0]);
    c[0][1] = WMMA_F16(a0, b1, c[0][1]);
    c[1][0] = WMMA_F16(a1, b0, c[1][0]);
    c[1][1] = WMMA_F16(a1, b1, c[1][1]);
  }
}

// Q projection: qb (f16) [S][E] laid out [s][h*64+d]
__global__ __launch_bounds__(256) void gemm_q_kernel(const h16* __restrict__ xb,
                                                     const h16* __restrict__ WqT,
                                                     h16* __restrict__ qb) {
  int w = blockIdx.x * 8 + (threadIdx.x >> 5);
  int m0 = (w >> 6) * 32, n0 = (w & 63) * 32;      // 64 x 64 tiles
  v8f c[2][2];
  wave_gemm(xb, WqT, E_, m0, n0, c);
  int lane = threadIdx.x & 31;
  int mbase = (lane & 16) ? 8 : 0, ncol = lane & 15;
#pragma unroll
  for (int i = 0; i < 2; i++)
#pragma unroll
    for (int j = 0; j < 2; j++)
#pragma unroll
      for (int r = 0; r < 8; r++) {
        int m = m0 + i * 16 + mbase + r, n = n0 + j * 16 + ncol;
        qb[(size_t)m * E_ + n] = (h16)c[i][j][r];
      }
}

// K/V projection: fp32 -> new_k/new_v (t >= CACHE), f16 -> Kb (mode 0) or Vt (mode 1)
__global__ __launch_bounds__(256) void gemm_kv_kernel(const h16* __restrict__ xb,
                                                      const h16* __restrict__ WT,
                                                      float* __restrict__ newkv,
                                                      h16* __restrict__ hdst, int mode) {
  int w = blockIdx.x * 8 + (threadIdx.x >> 5);
  int m0 = (w >> 4) * 32, n0 = (w & 15) * 32;      // 64 x 16 tiles (N = 512)
  v8f c[2][2];
  wave_gemm(xb, WT, E_, m0, n0, c);
  int lane = threadIdx.x & 31;
  int mbase = (lane & 16) ? 8 : 0, ncol = lane & 15;
#pragma unroll
  for (int i = 0; i < 2; i++)
#pragma unroll
    for (int j = 0; j < 2; j++)
#pragma unroll
      for (int r = 0; r < 8; r++) {
        int m = m0 + i * 16 + mbase + r, n = n0 + j * 16 + ncol;
        float val = c[i][j][r];
        int t = CACHE_ + m;
        newkv[(size_t)t * 512 + n] = val;
        if (mode == 0)
          hdst[((size_t)(n >> 6) * T_ + t) * DH_ + (n & 63)] = (h16)val;
        else
          hdst[(size_t)n * T_ + t] = (h16)val;
      }
}

// Output projection -> fp32 d_out
__global__ __launch_bounds__(256) void gemm_o_kernel(const h16* __restrict__ ab,
                                                     const h16* __restrict__ WoT,
                                                     float* __restrict__ out) {
  int w = blockIdx.x * 8 + (threadIdx.x >> 5);
  int m0 = (w >> 6) * 32, n0 = (w & 63) * 32;
  v8f c[2][2];
  wave_gemm(ab, WoT, E_, m0, n0, c);
  int lane = threadIdx.x & 31;
  int mbase = (lane & 16) ? 8 : 0, ncol = lane & 15;
#pragma unroll
  for (int i = 0; i < 2; i++)
#pragma unroll
    for (int j = 0; j < 2; j++)
#pragma unroll
      for (int r = 0; r < 8; r++) {
        int m = m0 + i * 16 + mbase + r, n = n0 + j * 16 + ncol;
        out[(size_t)m * E_ + n] = c[i][j][r];
      }
}

// ---------------- flash attention (one wave = one head x 16-query tile) ----------------
// Reductions over the 16 lanes of a half-wave (matches C-layout: a row lives in
// one VGPR across one 16-lane half; xor masks 1..8 never cross the half boundary).
__device__ __forceinline__ float half16_max(float v) {
#pragma unroll
  for (int m = 1; m < 16; m <<= 1) v = fmaxf(v, __shfl_xor(v, m, 32));
  return v;
}
__device__ __forceinline__ float half16_sum(float v) {
#pragma unroll
  for (int m = 1; m < 16; m <<= 1) v += __shfl_xor(v, m, 32);
  return v;
}

__global__ __launch_bounds__(256) void attn_kernel(const h16* __restrict__ qb,
                                                   const h16* __restrict__ Kb,
                                                   const h16* __restrict__ Vt,
                                                   h16* __restrict__ aout) {
  __shared__ h16 pls[8][16 * 64];                  // per-wave P staging (2 KB each)
  int wave = threadIdx.x >> 5;
  int lane = threadIdx.x & 31;
  int w  = blockIdx.x * 8 + wave;
  int h  = w >> 7;                                 // 128 query tiles per head
  int q0 = (w & 127) << 4;
  int hkv = h >> 2;                                // G = 4
  const h16* Qh  = qb + h * DH_;
  const h16* KbH = Kb + (size_t)hkv * T_ * DH_;
  const h16* VtH = Vt + (size_t)hkv * DH_ * T_;

  v16h qa0 = load_a_frag(Qh, E_, q0, 0);           // d = 0..31
  v16h qa1 = load_a_frag(Qh, E_, q0, 32);          // d = 32..63

  v8f o0 = {}, o1 = {}, o2 = {}, o3 = {};
  float mrow[8], lrow[8];                          // running max/sum (log2 domain)
#pragma unroll
  for (int r = 0; r < 8; r++) { mrow[r] = -1.0e30f; lrow[r] = 0.0f; }

  int mbase = (lane & 16) ? 8 : 0;
  int ncol  = lane & 15;
  h16* myP  = pls[wave];
  int tEnd  = (CACHE_ + q0 + 16 + 63) & ~63;       // <= T_ always (4080 -> 4096)

  for (int t0 = 0; t0 < tEnd; t0 += 64) {
    // ---- scores: four 16x16 tiles over 64 kv, K = D = 64 split into 2 x K32 ----
    v8f s[4];
#pragma unroll
    for (int j = 0; j < 4; j++) {
      v8f z = {};
      z = WMMA_F16(qa0, load_b_frag(KbH, DH_, t0 + 16 * j, 0),  z);
      z = WMMA_F16(qa1, load_b_frag(KbH, DH_, t0 + 16 * j, 32), z);
      s[j] = z;
    }

    // wave-uniform: does any element of this 64-col chunk cross the diagonal?
    bool needMask = (t0 + 63 > CACHE_ + q0);

    float alpha[8];
#pragma unroll
    for (int r = 0; r < 8; r++) {
      // convert to log2 domain (scale * log2e folded into one multiply)
      float e0 = s[0][r] * K2_, e1 = s[1][r] * K2_;
      float e2 = s[2][r] * K2_, e3 = s[3][r] * K2_;
      if (needMask) {
        int lim = CACHE_ + q0 + mbase + r;         // kv limit for this row
        if (t0      + ncol > lim) e0 = -1.0e30f;
        if (t0 + 16 + ncol > lim) e1 = -1.0e30f;
        if (t0 + 32 + ncol > lim) e2 = -1.0e30f;
        if (t0 + 48 + ncol > lim) e3 = -1.0e30f;
      }
      float cm = half16_max(fmaxf(fmaxf(e0, e1), fmaxf(e2, e3)));
      float mn = fmaxf(mrow[r], cm);
      alpha[r] = exp2f(mrow[r] - mn);
      e0 = exp2f(e0 - mn); e1 = exp2f(e1 - mn);
      e2 = exp2f(e2 - mn); e3 = exp2f(e3 - mn);
      float cs = half16_sum((e0 + e1) + (e2 + e3));
      lrow[r] = lrow[r] * alpha[r] + cs;
      mrow[r] = mn;
      s[0][r] = e0; s[1][r] = e1; s[2][r] = e2; s[3][r] = e3;
    }

    // rescale accumulators + stage P (C-layout -> row-major [16][64] f16 in LDS)
#pragma unroll
    for (int r = 0; r < 8; r++) {
      o0[r] *= alpha[r]; o1[r] *= alpha[r]; o2[r] *= alpha[r]; o3[r] *= alpha[r];
      h16* row = myP + (mbase + r) * 64 + ncol;
      row[0]  = (h16)s[0][r];
      row[16] = (h16)s[1][r];
      row[32] = (h16)s[2][r];
      row[48] = (h16)s[3][r];
    }

    // reload P as two 16x32 A operands (same-wave LDS ops are in-order; no barrier)
    const h16* pp = myP + (lane & 15) * 64 + ((lane & 16) ? 8 : 0);
    v16h pa0 = frag2(pp,      pp + 16);            // kv t0    .. t0+31
    v16h pa1 = frag2(pp + 32, pp + 48);            // kv t0+32 .. t0+63
    o0 = WMMA_F16(pa0, load_b_frag(VtH, T_, 0,  t0),      o0);
    o0 = WMMA_F16(pa1, load_b_frag(VtH, T_, 0,  t0 + 32), o0);
    o1 = WMMA_F16(pa0, load_b_frag(VtH, T_, 16, t0),      o1);
    o1 = WMMA_F16(pa1, load_b_frag(VtH, T_, 16, t0 + 32), o1);
    o2 = WMMA_F16(pa0, load_b_frag(VtH, T_, 32, t0),      o2);
    o2 = WMMA_F16(pa1, load_b_frag(VtH, T_, 32, t0 + 32), o2);
    o3 = WMMA_F16(pa0, load_b_frag(VtH, T_, 48, t0),      o3);
    o3 = WMMA_F16(pa1, load_b_frag(VtH, T_, 48, t0 + 32), o3);
  }

  // normalize and store f16 attn output [s][h*64+d]
#pragma unroll
  for (int r = 0; r < 8; r++) {
    float inv = 1.0f / lrow[r];
    int row = q0 + mbase + r;
    h16* dst = aout + (size_t)row * E_ + h * DH_;
    dst[ 0 + ncol] = (h16)(o0[r] * inv);
    dst[16 + ncol] = (h16)(o1[r] * inv);
    dst[32 + ncol] = (h16)(o2[r] * inv);
    dst[48 + ncol] = (h16)(o3[r] * inv);
  }
}

// ---------------- host-side launch ----------------
extern "C" void kernel_launch(void* const* d_in, const int* in_sizes, int n_in,
                              void* d_out, int out_size, void* d_ws, size_t ws_size,
                              hipStream_t stream) {
  const float* x      = (const float*)d_in[0];
  // d_in[1] = mask: pattern known analytically, ignored
  const float* prev_k = (const float*)d_in[2];
  const float* prev_v = (const float*)d_in[3];
  const float* Wq     = (const float*)d_in[4];
  const float* Wk     = (const float*)d_in[5];
  const float* Wv     = (const float*)d_in[6];
  const float* Wo     = (const float*)d_in[7];

  float* outp = (float*)d_out;                       // [S][E]
  float* newk = outp + (size_t)S_ * E_;              // [T][HKV*DH]
  float* newv = newk + (size_t)T_ * HKV_ * DH_;

  // workspace carve-up (f16), ~52 MB total
  h16* xb   = (h16*)d_ws;
  h16* WqT  = xb  + (size_t)S_ * E_;
  h16* WkT  = WqT + (size_t)E_ * E_;
  h16* WvT  = WkT + (size_t)E_ * 512;
  h16* WoT  = WvT + (size_t)E_ * 512;
  h16* qb   = WoT + (size_t)E_ * E_;
  h16* Kb   = qb  + (size_t)S_ * E_;
  h16* Vt   = Kb  + (size_t)HKV_ * T_ * DH_;
  h16* aout = Vt  + (size_t)HKV_ * DH_ * T_;

  cast_f16_kernel      <<<512, 256, 0, stream>>>(x, xb, S_ * E_);
  transpose_cast_kernel<<<512, 256, 0, stream>>>(Wq, WqT, E_, E_);
  transpose_cast_kernel<<<256, 256, 0, stream>>>(Wk, WkT, E_, 512);
  transpose_cast_kernel<<<256, 256, 0, stream>>>(Wv, WvT, E_, 512);
  transpose_cast_kernel<<<512, 256, 0, stream>>>(Wo, WoT, E_, E_);
  prep_cache_kernel    <<<512, 256, 0, stream>>>(prev_k, prev_v, newk, newv, Kb, Vt);

  gemm_q_kernel <<<512, 256, 0, stream>>>(xb, WqT, qb);           // 4096 waves
  gemm_kv_kernel<<<128, 256, 0, stream>>>(xb, WkT, newk, Kb, 0);  // 1024 waves
  gemm_kv_kernel<<<128, 256, 0, stream>>>(xb, WvT, newv, Vt, 1);
  attn_kernel   <<<512, 256, 0, stream>>>(qb, Kb, Vt, aout);      // 32 heads x 128 tiles
  gemm_o_kernel <<<512, 256, 0, stream>>>(aout, WoT, outp);
}
